// FourierUnit_10892037062921
// MI455X (gfx1250) — matrix operand order repaired and checked by reading, not compiled, not run on hardware
//
#include <hip/hip_runtime.h>
#include <hip/hip_bf16.h>
#include <math.h>

// ---------------------------------------------------------------------------
// Types for CDNA5 WMMA (wave32): v_wmma_f32_16x16x32_bf16
// ---------------------------------------------------------------------------
typedef __attribute__((ext_vector_type(16))) __bf16 v16bf;
typedef __attribute__((ext_vector_type(8)))  float  v8f;

union FragU {
  v16bf v;
  unsigned int u[8];
};

__device__ __forceinline__ v8f wmma_bf16(v16bf a, v16bf b, v8f c) {
#if defined(__HIP_DEVICE_COMPILE__)
  return __builtin_amdgcn_wmma_f32_16x16x32_bf16(false, a, false, b, (short)0, c,
                                                 false, false);
#else
  (void)a; (void)b;
  return c;
#endif
}

__device__ __forceinline__ unsigned short f2bf_bits(float f) {
  union { float f; unsigned int u; } a;
  a.f = f;
  unsigned int u = a.u;
  unsigned int r = u + 0x7FFFu + ((u >> 16) & 1u);  // round-to-nearest-even
  return (unsigned short)(r >> 16);
}

// ---------------------------------------------------------------------------
// Elementwise helpers
// ---------------------------------------------------------------------------
__global__ void r2c_kernel(const float* __restrict__ in, float2* __restrict__ out, int n) {
  int i = blockIdx.x * blockDim.x + threadIdx.x;
  if (i < n) { float2 v; v.x = in[i]; v.y = 0.f; out[i] = v; }
}

__global__ void magpha2c_kernel(const float* __restrict__ mag, const float* __restrict__ pha,
                                float2* __restrict__ out, int n) {
  int i = blockIdx.x * blockDim.x + threadIdx.x;
  if (i < n) {
    float m = mag[i], p = pha[i];
    float s, c;
    sincosf(p, &s, &c);
    float2 v; v.x = m * c; v.y = m * s;
    out[i] = v;
  }
}

__global__ void cabs_kernel(const float2* __restrict__ in, float* __restrict__ out, int n) {
  int i = blockIdx.x * blockDim.x + threadIdx.x;
  if (i < n) {
    float2 z = in[i];
    out[i] = sqrtf(z.x * z.x + z.y * z.y);
  }
}

// ---------------------------------------------------------------------------
// Batched radix-2 LDS FFT, one line (row or column) per block.
// dirsign = -1: forward (e^{-i...}); +1: inverse (caller applies 1/(H*W) scale)
// rows: base = line*N, stride 1 ; cols: base = img*H*W + col, stride W (N==H)
// ---------------------------------------------------------------------------
__global__ void fft1d_kernel(const float2* __restrict__ in, float2* __restrict__ out,
                             int N, int logN, int dirsign, float scale,
                             int is_col, int H, int W) {
  __shared__ float2 lds[128];
  const int line = blockIdx.x;
  const int tid = threadIdx.x;
  long base;
  int stride;
  if (is_col) {
    int img = line / W;
    int col = line - img * W;
    base = (long)img * H * W + col;
    stride = W;
  } else {
    base = (long)line * N;
    stride = 1;
  }
  for (int i = tid; i < N; i += blockDim.x) {
    int r = (int)(__brev((unsigned)i) >> (32 - logN));
    lds[r] = in[base + (long)i * stride];
  }
  __syncthreads();
  for (int len = 2; len <= N; len <<= 1) {
    int halfl = len >> 1;
    for (int t = tid; t < (N >> 1); t += blockDim.x) {
      int grp = t / halfl;
      int j = t - grp * halfl;
      int idx = grp * len + j;
      float ang = (float)dirsign * 6.28318530717958647692f * (float)j / (float)len;
      float s, c;
      sincosf(ang, &s, &c);
      float2 u = lds[idx];
      float2 w = lds[idx + halfl];
      float2 wv;
      wv.x = w.x * c - w.y * s;
      wv.y = w.x * s + w.y * c;
      float2 a, b;
      a.x = u.x + wv.x; a.y = u.y + wv.y;
      b.x = u.x - wv.x; b.y = u.y - wv.y;
      lds[idx] = a;
      lds[idx + halfl] = b;
    }
    __syncthreads();
  }
  for (int i = tid; i < N; i += blockDim.x) {
    float2 v = lds[i];
    v.x *= scale; v.y *= scale;
    out[base + (long)i * stride] = v;
  }
}

// ---------------------------------------------------------------------------
// Repack a (K x N) row-major f32 weight into bf16 wave32 B-fragments.
// Fragment tile = 32(K) x 16(N). B layout (dense bf16 32x16, contiguous k order of this code):
//   lanes 0-15 hold K=0..15 (col = lane), lanes 16-31 hold K=16..31;
//   VGPR v of a lane holds K-pair {16*half + 2v, 16*half + 2v + 1}.
// frag[((kt*NT + nt)*32 + lane)*8 + v], zero-padded past K/N.
// ---------------------------------------------------------------------------
__global__ __launch_bounds__(32)
void repack_b_kernel(const float* __restrict__ W, int K, int N, int NT,
                     unsigned int* __restrict__ frag) {
  const int lane = threadIdx.x;
  const int t = blockIdx.x;
  const int kt = t / NT, nt = t - kt * NT;
  const int n = nt * 16 + (lane & 15);
  const int half = lane >> 4;
#pragma unroll
  for (int v = 0; v < 8; ++v) {
    int k = kt * 32 + half * 16 + 2 * v;
    float lo = (k < K && n < N) ? W[k * N + n] : 0.f;
    float hi = (k + 1 < K && n < N) ? W[(k + 1) * N + n] : 0.f;
    unsigned int p = (unsigned int)f2bf_bits(lo) |
                     ((unsigned int)f2bf_bits(hi) << 16);
    frag[(t * 32 + lane) * 8 + v] = p;
  }
}

// ---------------------------------------------------------------------------
// Fused LIIF query kernel (one branch: mode 0 = magnitude, 1 = phase).
// 4 waves (128 threads) per block, 16 queries per block, M = 64 rows
// (row = offset*16 + query) so every B-fragment load feeds 4 WMMAs.
// The nt loop is split across the 4 waves (disjoint B-fragments, shared LDS
// activation tiles).
// A-fragment layout (16-bit A 16x32, contiguous k, same order as B):
//   lane: m = lane%16, half = lane/16; VGPR v holds K-pair
//   { 16*half + 2v, +1 } (k = 16*half + s)
//   => u32 pair index = v + 8*half + kt*16.
// C/D: acc[r] = C[m = r + 8*half][n = lane%16].
// ---------------------------------------------------------------------------
#define QM 16

__global__ __launch_bounds__(128)
void query_freq_kernel(const float2* __restrict__ cfeat,   // (8,128,64,64) complex
                       const float2* __restrict__ cguide,  // (8,128,128,128) complex
                       const float* __restrict__ coord,    // (8,16384,2)
                       const unsigned int* __restrict__ fw1, const float* __restrict__ b1,
                       const unsigned int* __restrict__ fw2, const float* __restrict__ b2,
                       const unsigned int* __restrict__ fw3, const float* __restrict__ b3,
                       float* __restrict__ outp,           // (8,32,16384)
                       int mode) {
  __shared__ __align__(16) unsigned short sAct[4 * QM][288];   // 258 used, pad 288
  __shared__ __align__(16) unsigned short sHid1[4 * QM][256];
  __shared__ __align__(16) unsigned short sHid2[4 * QM][128];
  __shared__ float sPred[4][QM][33];
  __shared__ float sCy[QM], sCx[QM];
  __shared__ int sB[QM], sN[QM];

  const int tid = threadIdx.x;
  const int lane = tid & 31;
  const int w = tid >> 5;       // wave id, 0..3
  const int m = lane & 15;
  const int half = lane >> 4;

  if (tid < QM) {
    int gq = blockIdx.x * QM + tid;
    int bidx = gq >> 14;        // / 16384
    int n = gq & 16383;
    sB[tid] = bidx;
    sN[tid] = n;
    sCy[tid] = coord[((long)bidx * 16384 + n) * 2 + 0];
    sCx[tid] = coord[((long)bidx * 16384 + n) * 2 + 1];
  }
  // Zero K-padding region (cols 256..287) of all 64 rows; 256/257 overwritten.
  for (int i = tid; i < 64 * 32; i += 128)
    sAct[i >> 5][256 + (i & 31)] = 0;
  __syncthreads();

  // Guide gather (cols 128..255) — identical for all 4 offsets (raw coord),
  // replicate into all 4 row blocks.
  for (int i = tid; i < QM * 128; i += 128) {
    int q = i >> 7, c = i & 127;
    float cy = sCy[q], cx = sCx[q];
    float fy = rintf(((cy + 1.f) * 128.f - 1.f) * 0.5f);
    float fx = rintf(((cx + 1.f) * 128.f - 1.f) * 0.5f);
    float val = 0.f;
    if (fy >= 0.f && fy < 128.f && fx >= 0.f && fx < 128.f) {
      int iy = (int)fy, ix = (int)fx;
      float2 z = cguide[(((long)sB[q] * 128 + c) * 128 + iy) * 128 + ix];
      val = mode ? atan2f(z.y, z.x) : sqrtf(z.x * z.x + z.y * z.y);
    }
    unsigned short bv = f2bf_bits(val);
    sAct[q][128 + c] = bv;
    sAct[QM + q][128 + c] = bv;
    sAct[2 * QM + q][128 + c] = bv;
    sAct[3 * QM + q][128 + c] = bv;
  }

  // Feat gather (cols 0..127) + relative coords (cols 256,257), all 4 offsets.
  for (int i = tid; i < 4 * QM * 128; i += 128) {
    int off = i >> 11;           // / 2048
    int j = i & 2047;
    int q = j >> 7, c = j & 127;
    float vx = (off & 2) ? 1.f : -1.f;   // reference loop order: vx outer
    float vy = (off & 1) ? 1.f : -1.f;
    float cy = sCy[q] + vx * (1.f / 64.f);
    float cx = sCx[q] + vy * (1.f / 64.f);
    float fy = rintf(((cy + 1.f) * 64.f - 1.f) * 0.5f);
    float fx = rintf(((cx + 1.f) * 64.f - 1.f) * 0.5f);
    bool valid = (fy >= 0.f && fy < 64.f && fx >= 0.f && fx < 64.f);
    float val = 0.f;
    int iy = 0, ix = 0;
    if (valid) {
      iy = (int)fy; ix = (int)fx;
      float2 z = cfeat[(((long)sB[q] * 128 + c) * 64 + iy) * 64 + ix];
      val = mode ? atan2f(z.y, z.x) : sqrtf(z.x * z.x + z.y * z.y);
    }
    int row = off * QM + q;
    sAct[row][c] = f2bf_bits(val);
    if (c == 0) {
      float qy = valid ? (-1.f + (2.f * (float)iy + 1.f) * (1.f / 64.f)) : 0.f;
      float qx = valid ? (-1.f + (2.f * (float)ix + 1.f) * (1.f / 64.f)) : 0.f;
      sAct[row][256] = f2bf_bits((sCy[q] - qy) * 64.f);
      sAct[row][257] = f2bf_bits((sCx[q] - qx) * 64.f);
    }
  }
  __syncthreads();

  // ---- Layer 1: [64 x 288] @ [288 x 256], bias + ReLU -> sHid1 -------------
  for (int nt = w; nt < 16; nt += 4) {
    v8f acc[4];
#pragma unroll
    for (int mt = 0; mt < 4; ++mt)
#pragma unroll
      for (int r = 0; r < 8; ++r) acc[mt][r] = 0.f;
    for (int kt = 0; kt < 9; ++kt) {
      FragU B;
      const unsigned int* bp = &fw1[((kt * 16 + nt) * 32 + lane) * 8];
#pragma unroll
      for (int v = 0; v < 8; ++v) B.u[v] = bp[v];
      int abase = kt * 16 + half * 8;
#pragma unroll
      for (int mt = 0; mt < 4; ++mt) {
        FragU A;
        const unsigned int* arow = (const unsigned int*)&sAct[mt * QM + m][0];
#pragma unroll
        for (int v = 0; v < 8; ++v) A.u[v] = arow[abase + v];
        acc[mt] = wmma_bf16(A.v, B.v, acc[mt]);
      }
    }
    float bias = b1[nt * 16 + m];
#pragma unroll
    for (int mt = 0; mt < 4; ++mt)
#pragma unroll
      for (int r = 0; r < 8; ++r) {
        float x = acc[mt][r] + bias;
        x = x > 0.f ? x : 0.f;
        sHid1[mt * QM + r + 8 * half][nt * 16 + m] = f2bf_bits(x);
      }
  }
  __syncthreads();

  // ---- Layer 2: [64 x 256] @ [256 x 128], bias + ReLU -> sHid2 -------------
  for (int nt = w; nt < 8; nt += 4) {
    v8f acc[4];
#pragma unroll
    for (int mt = 0; mt < 4; ++mt)
#pragma unroll
      for (int r = 0; r < 8; ++r) acc[mt][r] = 0.f;
    for (int kt = 0; kt < 8; ++kt) {
      FragU B;
      const unsigned int* bp = &fw2[((kt * 8 + nt) * 32 + lane) * 8];
#pragma unroll
      for (int v = 0; v < 8; ++v) B.u[v] = bp[v];
      int abase = kt * 16 + half * 8;
#pragma unroll
      for (int mt = 0; mt < 4; ++mt) {
        FragU A;
        const unsigned int* arow = (const unsigned int*)&sHid1[mt * QM + m][0];
#pragma unroll
        for (int v = 0; v < 8; ++v) A.u[v] = arow[abase + v];
        acc[mt] = wmma_bf16(A.v, B.v, acc[mt]);
      }
    }
    float bias = b2[nt * 16 + m];
#pragma unroll
    for (int mt = 0; mt < 4; ++mt)
#pragma unroll
      for (int r = 0; r < 8; ++r) {
        float x = acc[mt][r] + bias;
        x = x > 0.f ? x : 0.f;
        sHid2[mt * QM + r + 8 * half][nt * 16 + m] = f2bf_bits(x);
      }
  }
  __syncthreads();

  // ---- Layer 3: [64 x 128] @ [128 x 48(33)], bias -> sPred -----------------
  if (w < 3) {
    int nt = w;
    v8f acc[4];
#pragma unroll
    for (int mt = 0; mt < 4; ++mt)
#pragma unroll
      for (int r = 0; r < 8; ++r) acc[mt][r] = 0.f;
    for (int kt = 0; kt < 4; ++kt) {
      FragU B;
      const unsigned int* bp = &fw3[((kt * 3 + nt) * 32 + lane) * 8];
#pragma unroll
      for (int v = 0; v < 8; ++v) B.u[v] = bp[v];
      int abase = kt * 16 + half * 8;
#pragma unroll
      for (int mt = 0; mt < 4; ++mt) {
        FragU A;
        const unsigned int* arow = (const unsigned int*)&sHid2[mt * QM + m][0];
#pragma unroll
        for (int v = 0; v < 8; ++v) A.u[v] = arow[abase + v];
        acc[mt] = wmma_bf16(A.v, B.v, acc[mt]);
      }
    }
    int col = nt * 16 + m;
    if (col < 33) {
      float bias = b3[col];
#pragma unroll
      for (int mt = 0; mt < 4; ++mt)
#pragma unroll
        for (int r = 0; r < 8; ++r)
          sPred[mt][r + 8 * half][col] = acc[mt][r] + bias;
    }
  }
  __syncthreads();

  // Softmax over the 4 offsets (logit = channel 32) + weighted sum of 0..31.
  for (int i = tid; i < QM * 32; i += 128) {
    int q = i >> 5, ch = i & 31;
    float l0 = sPred[0][q][32], l1 = sPred[1][q][32];
    float l2 = sPred[2][q][32], l3 = sPred[3][q][32];
    float mx = fmaxf(fmaxf(l0, l1), fmaxf(l2, l3));
    float e0 = __expf(l0 - mx), e1 = __expf(l1 - mx);
    float e2 = __expf(l2 - mx), e3 = __expf(l3 - mx);
    float inv = 1.f / (e0 + e1 + e2 + e3);
    float val = (sPred[0][q][ch] * e0 + sPred[1][q][ch] * e1 +
                 sPred[2][q][ch] * e2 + sPred[3][q][ch] * e3) * inv;
    outp[((long)sB[q] * 32 + ch) * 16384 + sN[q]] = val;
  }
}

// ---------------------------------------------------------------------------
// Launcher.  Workspace layout (floats):
//   [0)              8,388,608  cfeat  : feat FFT, complex (8,128,64,64)
//   [8388608)       33,554,432  cguide : guide FFT, complex (8,128,128,128)
//                                         (reused as ifft complex buffer)
//   [41943040)       4,194,304  magout : (8,32,16384)
//   [46137344)       4,194,304  phaout : (8,32,16384)
//   [50331648)         112,640  bf16 weight fragments (as u32)
// Total ~202 MB.
// ---------------------------------------------------------------------------
extern "C" void kernel_launch(void* const* d_in, const int* in_sizes, int n_in,
                              void* d_out, int out_size, void* d_ws, size_t ws_size,
                              hipStream_t stream) {
  (void)in_sizes; (void)n_in; (void)out_size; (void)ws_size;
  const float* feat  = (const float*)d_in[0];
  const float* coord = (const float*)d_in[1];
  const float* guide = (const float*)d_in[2];
  const float* a_w1 = (const float*)d_in[3];  const float* a_b1 = (const float*)d_in[4];
  const float* a_w2 = (const float*)d_in[5];  const float* a_b2 = (const float*)d_in[6];
  const float* a_w3 = (const float*)d_in[7];  const float* a_b3 = (const float*)d_in[8];
  const float* p_w1 = (const float*)d_in[9];  const float* p_b1 = (const float*)d_in[10];
  const float* p_w2 = (const float*)d_in[11]; const float* p_b2 = (const float*)d_in[12];
  const float* p_w3 = (const float*)d_in[13]; const float* p_b3 = (const float*)d_in[14];
  float* out = (float*)d_out;

  float* ws = (float*)d_ws;
  float2* cfeat  = (float2*)ws;
  float2* cguide = (float2*)(ws + 8388608);
  float*  magout = ws + 41943040;
  float*  phaout = ws + 46137344;
  unsigned int* frag = (unsigned int*)(ws + 50331648);
  unsigned int* fa1 = frag;
  unsigned int* fa2 = fa1 + 36864;
  unsigned int* fa3 = fa2 + 16384;
  unsigned int* fp1 = fa3 + 3072;
  unsigned int* fp2 = fp1 + 36864;
  unsigned int* fp3 = fp2 + 16384;
  float2* cout = cguide;   // guide spectrum dead after query kernels

  // Forward FFTs (rows then cols, in place per line).
  r2c_kernel<<<16384, 256, 0, stream>>>(feat, cfeat, 4194304);
  fft1d_kernel<<<65536, 32, 0, stream>>>(cfeat, cfeat, 64, 6, -1, 1.f, 0, 64, 64);
  fft1d_kernel<<<65536, 32, 0, stream>>>(cfeat, cfeat, 64, 6, -1, 1.f, 1, 64, 64);

  r2c_kernel<<<65536, 256, 0, stream>>>(guide, cguide, 16777216);
  fft1d_kernel<<<131072, 64, 0, stream>>>(cguide, cguide, 128, 7, -1, 1.f, 0, 128, 128);
  fft1d_kernel<<<131072, 64, 0, stream>>>(cguide, cguide, 128, 7, -1, 1.f, 1, 128, 128);

  // Pre-swizzle MLP weights into bf16 WMMA B-fragments.
  repack_b_kernel<<<144, 32, 0, stream>>>(a_w1, 258, 256, 16, fa1);
  repack_b_kernel<<< 64, 32, 0, stream>>>(a_w2, 256, 128,  8, fa2);
  repack_b_kernel<<< 12, 32, 0, stream>>>(a_w3, 128,  33,  3, fa3);
  repack_b_kernel<<<144, 32, 0, stream>>>(p_w1, 258, 256, 16, fp1);
  repack_b_kernel<<< 64, 32, 0, stream>>>(p_w2, 256, 128,  8, fp2);
  repack_b_kernel<<< 12, 32, 0, stream>>>(p_w3, 128,  33,  3, fp3);

  // Fused gather + WMMA MLP + softmax combine (mag branch, then phase branch).
  query_freq_kernel<<<8192, 128, 0, stream>>>(cfeat, cguide, coord,
                                              fa1, a_b1, fa2, a_b2, fa3, a_b3,
                                              magout, 0);
  query_freq_kernel<<<8192, 128, 0, stream>>>(cfeat, cguide, coord,
                                              fp1, p_b1, fp2, p_b2, fp3, p_b3,
                                              phaout, 1);

  // Recombine, inverse FFT (scale 1/16384 on the final pass), magnitude out.
  magpha2c_kernel<<<16384, 256, 0, stream>>>(magout, phaout, cout, 4194304);
  fft1d_kernel<<<32768, 64, 0, stream>>>(cout, cout, 128, 7, +1, 1.f, 0, 128, 128);
  fft1d_kernel<<<32768, 64, 0, stream>>>(cout, cout, 128, 7, +1, 1.f / 16384.f, 1, 128, 128);
  cabs_kernel<<<16384, 256, 0, stream>>>(cout, out, 4194304);
}
